// ContinuousAttentiveStatisticsPooling_87007447483014
// MI455X (gfx1250) — matrix operand order repaired and checked by reading, not compiled, not run on hardware
//
#include <hip/hip_runtime.h>
#include <hip/hip_bf16.h>
#include <stdint.h>

// Problem constants (from reference): B=8, C=256, L=8192, A=128
#define BB 8
#define CC 256
#define LL 8192
#define AA 128
#define EPSF 1e-12f
#define NEGS -3.0e38f     // sentinel "-inf" that keeps exp() arithmetic finite
#define NBLK (LL / 128)   // 64 L-tiles per batch

typedef __bf16 v16bf __attribute__((ext_vector_type(16)));
typedef float  v8f   __attribute__((ext_vector_type(8)));

union FragU {
    uint4 q[2];   // 32 bytes
    v16bf h;      // 16 bf16
};

__device__ __forceinline__ unsigned short f32_to_bf16_rne(float f) {
    uint32_t u = __float_as_uint(f);
    u += 0x7FFFu + ((u >> 16) & 1u);   // round-to-nearest-even
    return (unsigned short)(u >> 16);
}

// Merge two online-softmax partial tuples (m, Z, sv, svv).
// With the NEGS sentinel both-empty merges stay finite (exp(0)=1 scaled by Z=0).
__device__ __forceinline__ void online_combine(float& m, float& Z, float& sv, float& svv,
                                               float m2, float Z2, float sv2, float svv2) {
    float mn = fmaxf(m, m2);
    float f1 = expf(m - mn);
    float f2 = expf(m2 - mn);
    Z   = Z * f1 + Z2 * f2;
    sv  = sv * f1 + sv2 * f2;
    svv = svv * f1 + svv2 * f2;
    m   = mn;
}

// ---------------------------------------------------------------------------
// Kernel 1: masked global mean / std per (b, c).  One block per (c, b).
// Runs first so x is hot in the 192MB L2 for the transpose kernel.
// ---------------------------------------------------------------------------
__global__ __launch_bounds__(256)
void k_stats(const float* __restrict__ x, const float* __restrict__ lengths,
             float* __restrict__ gmean, float* __restrict__ gstd) {
    int c = blockIdx.x, b = blockIdx.y, t = threadIdx.x;
    const float* row = x + ((size_t)b * CC + c) * LL;
    float tlen = lengths[b] * (float)LL;
    float s1 = 0.f, s2 = 0.f;
    for (int l = t; l < LL; l += 256) {
        if ((float)l < tlen) { float v = row[l]; s1 += v; s2 += v * v; }
    }
    __shared__ float r1[256], r2[256];
    r1[t] = s1; r2[t] = s2;
    __syncthreads();
    for (int off = 128; off > 0; off >>= 1) {
        if (t < off) { r1[t] += r1[t + off]; r2[t] += r2[t + off]; }
        __syncthreads();
    }
    if (t == 0) {
        int cnt = (int)ceilf(tlen); if (cnt > LL) cnt = LL;
        float inv = 1.f / (float)cnt;
        float m = r1[0] * inv;
        float var = r2[0] * inv - m * m;
        gmean[b * CC + c] = m;
        gstd[b * CC + c]  = sqrtf(fmaxf(var, EPSF));
    }
}

// ---------------------------------------------------------------------------
// Kernel 2: transpose + convert x[B,C,L] f32 -> xT[B,L,C] bf16.
// The 16c x 256l f32 tile is pulled into LDS with CDNA5 async copies
// (global_load_async_to_lds_b128, ASYNCcnt-tracked, no VGPR round trip),
// then converted/packed and stored as row-major bf16 (WMMA A-operand layout:
// fragments become two B128 loads per lane).
// ---------------------------------------------------------------------------
__global__ __launch_bounds__(256)
void k_transpose(const float* __restrict__ x, unsigned short* __restrict__ xT) {
    int lt = blockIdx.x;            // L / 256
    int ct = blockIdx.y;            // C / 16
    int b  = blockIdx.z;
    int t  = threadIdx.x;
    __shared__ __align__(16) float ldsf[16 * 256];   // [c-within-tile][l] 16KB

    // ---- async copy: 1024 x b128 per block; per wave each lane copies 16B,
    // consecutive lanes consecutive l -> 512B contiguous per wave issue.
    #pragma unroll
    for (int r = 0; r < 4; ++r) {
        int linear = r * 256 + t;           // 0..1023
        int j  = linear >> 6;               // channel within tile (16 rows)
        int l4 = (linear & 63) << 2;        // l offset, 4 floats per copy
        uint32_t lds_addr = (uint32_t)(uintptr_t)&ldsf[j * 256 + l4];
        const float* g = x + ((size_t)b * CC + ct * 16 + j) * LL + lt * 256 + l4;
        asm volatile("global_load_async_to_lds_b128 %0, %1, off"
                     :: "v"(lds_addr), "v"(g) : "memory");
    }
    asm volatile("s_wait_asynccnt 0x0" ::: "memory");
    __syncthreads();

    // ---- convert + pack: thread t owns row l = lt*256 + t, 16 bf16 = 32B
    int l = lt * 256 + t;
    __align__(16) unsigned short tmp[16];
    #pragma unroll
    for (int j = 0; j < 16; ++j) tmp[j] = f32_to_bf16_rne(ldsf[j * 256 + t]);
    uint4* dst = (uint4*)(xT + ((size_t)b * LL + l) * CC + ct * 16);
    dst[0] = *(const uint4*)&tmp[0];
    dst[1] = *(const uint4*)&tmp[8];
}

// ---------------------------------------------------------------------------
// Kernel 3: bf16 weight slices, row-major [n][k] (B-operand: 32 contiguous
// bytes per lane).
// ---------------------------------------------------------------------------
__global__ __launch_bounds__(256)
void k_wconv(const float* __restrict__ wv, const float* __restrict__ wt,
             const float* __restrict__ wc,
             unsigned short* __restrict__ wv_bf, unsigned short* __restrict__ wt_bf,
             unsigned short* __restrict__ wc_bf) {
    int i = blockIdx.x * 256 + threadIdx.x;
    if (i < CC * CC) { int n = i / CC, k = i % CC; wv_bf[i] = f32_to_bf16_rne(wv[(size_t)n * (3 * CC) + k]); }
    if (i < AA * CC) { int n = i / CC, k = i % CC; wt_bf[i] = f32_to_bf16_rne(wt[(size_t)n * (3 * CC) + k]); }
    if (i < CC * AA) { wc_bf[i] = f32_to_bf16_rne(wc[i]); }   // w_conv is (C,A) row-major
}

// ---------------------------------------------------------------------------
// Kernel 4: fold the gmean/gstd concat columns into per-batch f32 biases.
// ---------------------------------------------------------------------------
__global__ __launch_bounds__(256)
void k_bias(const float* __restrict__ gmean, const float* __restrict__ gstd,
            const float* __restrict__ w_val, const float* __restrict__ b_val,
            const float* __restrict__ w_tdnn, const float* __restrict__ b_tdnn,
            float* __restrict__ bias_val, float* __restrict__ bias_tdnn) {
    int b = blockIdx.x, t = threadIdx.x;
    __shared__ float gm[CC], gs[CC];
    gm[t] = gmean[b * CC + t];
    gs[t] = gstd[b * CC + t];
    __syncthreads();
    {
        const float* row = w_val + (size_t)t * (3 * CC);
        float acc = b_val[t];
        #pragma unroll 4
        for (int f = 0; f < CC; ++f) acc += gm[f] * row[CC + f] + gs[f] * row[2 * CC + f];
        bias_val[b * CC + t] = acc;
    }
    if (t < AA) {
        const float* row = w_tdnn + (size_t)t * (3 * CC);
        float acc = b_tdnn[t];
        #pragma unroll 4
        for (int f = 0; f < CC; ++f) acc += gm[f] * row[CC + f] + gs[f] * row[2 * CC + f];
        bias_tdnn[b * AA + t] = acc;
    }
}

// ---------------------------------------------------------------------------
// Kernel 5: fully fused WMMA GEMMs + online masked softmax statistics.
// Block = 8 waves, 128 L-rows of one batch; wave owns a 16-row tile.
//   GEMM2: H[16,128] = xT*Wt^T (+bias, relu, BN)  -> LDS bf16 (per-wave region)
//   main loop per 16-col tile nt:
//     accV = xT*Wv^T (+bias_val)   [values]
//     accS = H *Wc^T (+b_conv)     [scores]  (same D layout => same (l,c) pairs)
//     per-lane online tuple over 8 rows (length-masked),
//     partner-lane merge via ds_bpermute (lane^16 holds the same column),
//     lanes 0..15 store one float4 per column into LDS (reusing the H buffer).
//   cross-wave merge -> one float4 partial per (b, block, c).
// V and S never touch HBM.
// ---------------------------------------------------------------------------
__global__ __launch_bounds__(256)
void k_gemm(const unsigned short* __restrict__ xT,
            const unsigned short* __restrict__ wv_bf,
            const unsigned short* __restrict__ wt_bf,
            const unsigned short* __restrict__ wc_bf,
            const float* __restrict__ bias_val,
            const float* __restrict__ bias_tdnn,
            const float* __restrict__ bn_gamma, const float* __restrict__ bn_beta,
            const float* __restrict__ b_conv,
            const float* __restrict__ lengths,
            float4* __restrict__ partial) {
    int b    = blockIdx.y;
    int blk  = blockIdx.x;
    int wave = threadIdx.x >> 5;
    int lane = threadIdx.x & 31;
    int l0   = blk * 128 + wave * 16;
    int rrow  = lane & 15;          // A-operand row within tile
    int khalf = (lane >> 4) * 16;   // A-operand K-group offset
    int ncol  = lane & 15;          // B/D column
    int kb16  = (lane >> 4) * 16;   // B-operand K-group offset
    int mh    = (lane >> 4) * 8;    // D row group
    float tlen = lengths[b] * (float)LL;

    // 32KB: per-wave 4KB region used first as bf16 H tile (16x128), then
    // reused as 256 x float4 online-softmax partials (same wave only).
    __shared__ __align__(16) unsigned short hl[8 * 16 * AA];
    float4* part = (float4*)hl;     // part[wave*256 + c]

    // ---- A fragments of the 16x256 xT tile (8 K-steps), reused by GEMM1+2
    FragU a[8];
    const unsigned short* arow = xT + ((size_t)b * LL + l0 + rrow) * CC + khalf;
    #pragma unroll
    for (int ks = 0; ks < 8; ++ks) {
        a[ks].q[0] = *(const uint4*)(arow + ks * 32);
        a[ks].q[1] = *(const uint4*)(arow + ks * 32 + 8);
    }
    // Speculative prefetch of the next 128-row tile (-> global_prefetch_b8)
    __builtin_prefetch(arow + (size_t)128 * CC, 0, 1);

    // ---- GEMM2: TDNN -> relu -> BN -> bf16 H tile in this wave's LDS region
    #pragma unroll 1
    for (int nt = 0; nt < 8; ++nt) {
        v8f acc = {};
        const unsigned short* brow = wt_bf + (size_t)(nt * 16 + ncol) * CC + kb16;
        #pragma unroll
        for (int ks = 0; ks < 8; ++ks) {
            FragU bf;
            bf.q[0] = *(const uint4*)(brow + ks * 32);
            bf.q[1] = *(const uint4*)(brow + ks * 32 + 8);
            acc = __builtin_amdgcn_wmma_f32_16x16x32_bf16(false, a[ks].h, false, bf.h,
                                                          (short)0, acc, false, false);
        }
        int ach = nt * 16 + ncol;
        float bias = bias_tdnn[b * AA + ach];
        float gam = bn_gamma[ach], bet = bn_beta[ach];
        #pragma unroll
        for (int v = 0; v < 8; ++v) {
            float hv = fmaxf(acc[v] + bias, 0.f) * gam + bet;
            hl[wave * (16 * AA) + (mh + v) * AA + ach] = f32_to_bf16_rne(hv);
        }
    }

    // ---- H A-fragments from LDS (same wave wrote them; dscnt handled by compiler)
    FragU ah[4];
    const unsigned short* hrow = hl + wave * (16 * AA) + rrow * AA + khalf;
    #pragma unroll
    for (int ks = 0; ks < 4; ++ks) {
        ah[ks].q[0] = *(const uint4*)(hrow + ks * 32);
        ah[ks].q[1] = *(const uint4*)(hrow + ks * 32 + 8);
    }

    // ---- main loop: values + scores per 16-column tile, online statistics
    #pragma unroll 1
    for (int nt = 0; nt < 16; ++nt) {
        v8f accV = {};
        const unsigned short* bv = wv_bf + (size_t)(nt * 16 + ncol) * CC + kb16;
        #pragma unroll
        for (int ks = 0; ks < 8; ++ks) {
            FragU bf;
            bf.q[0] = *(const uint4*)(bv + ks * 32);
            bf.q[1] = *(const uint4*)(bv + ks * 32 + 8);
            accV = __builtin_amdgcn_wmma_f32_16x16x32_bf16(false, a[ks].h, false, bf.h,
                                                           (short)0, accV, false, false);
        }
        v8f accS = {};
        const unsigned short* bc = wc_bf + (size_t)(nt * 16 + ncol) * AA + kb16;
        #pragma unroll
        for (int ks = 0; ks < 4; ++ks) {
            FragU bf;
            bf.q[0] = *(const uint4*)(bc + ks * 32);
            bf.q[1] = *(const uint4*)(bc + ks * 32 + 8);
            accS = __builtin_amdgcn_wmma_f32_16x16x32_bf16(false, ah[ks].h, false, bf.h,
                                                           (short)0, accS, false, false);
        }
        float biasV = bias_val[b * CC + nt * 16 + ncol];
        float biasS = b_conv[nt * 16 + ncol];

        // per-lane masked online softmax over this lane's 8 rows
        float m = NEGS, Z = 0.f, sv = 0.f, svv = 0.f;
        #pragma unroll
        for (int v = 0; v < 8; ++v) {
            if ((float)(l0 + mh + v) < tlen) m = fmaxf(m, accS[v] + biasS);
        }
        #pragma unroll
        for (int v = 0; v < 8; ++v) {
            bool valid = (float)(l0 + mh + v) < tlen;
            float s   = accS[v] + biasS;
            float val = accV[v] + biasV;
            float e = valid ? expf(s - m) : 0.f;
            Z += e; sv += e * val; svv += e * val * val;
        }

        // partner lane (lane^16) holds the same column's other 8 rows
        int pidx = ((lane ^ 16) << 2);
        float m2   = __int_as_float(__builtin_amdgcn_ds_bpermute(pidx, __float_as_int(m)));
        float Z2   = __int_as_float(__builtin_amdgcn_ds_bpermute(pidx, __float_as_int(Z)));
        float sv2  = __int_as_float(__builtin_amdgcn_ds_bpermute(pidx, __float_as_int(sv)));
        float svv2 = __int_as_float(__builtin_amdgcn_ds_bpermute(pidx, __float_as_int(svv)));
        online_combine(m, Z, sv, svv, m2, Z2, sv2, svv2);

        if (lane < 16)
            part[wave * 256 + nt * 16 + lane] = make_float4(m, Z, sv, svv);
    }
    __syncthreads();

    // ---- cross-wave merge: one thread per channel
    {
        int c = threadIdx.x;
        float4 p = part[c];
        float m = p.x, Z = p.y, sv = p.z, svv = p.w;
        #pragma unroll
        for (int w = 1; w < 8; ++w) {
            float4 q = part[w * 256 + c];
            online_combine(m, Z, sv, svv, q.x, q.y, q.z, q.w);
        }
        partial[((size_t)b * NBLK + blk) * CC + c] = make_float4(m, Z, sv, svv);
    }
}

// ---------------------------------------------------------------------------
// Kernel 6: fold 64 block partials per (b,c) -> amean / astd.
// ---------------------------------------------------------------------------
__global__ __launch_bounds__(256)
void k_combine(const float4* __restrict__ partial, float* __restrict__ out) {
    int b = blockIdx.x, c = threadIdx.x;
    float m = NEGS, Z = 0.f, sv = 0.f, svv = 0.f;
    #pragma unroll 4
    for (int blk = 0; blk < NBLK; ++blk) {
        float4 q = partial[((size_t)b * NBLK + blk) * CC + c];
        online_combine(m, Z, sv, svv, q.x, q.y, q.z, q.w);
    }
    float amean = sv / Z;
    float avar  = svv / Z - amean * amean;
    out[b * (2 * CC) + c]      = amean;
    out[b * (2 * CC) + CC + c] = sqrtf(fmaxf(avar, EPSF));
}

// ---------------------------------------------------------------------------
extern "C" void kernel_launch(void* const* d_in, const int* in_sizes, int n_in,
                              void* d_out, int out_size, void* d_ws, size_t ws_size,
                              hipStream_t stream) {
    const float* x        = (const float*)d_in[0];
    const float* lengths  = (const float*)d_in[1];
    const float* w_val    = (const float*)d_in[2];
    const float* b_val    = (const float*)d_in[3];
    const float* w_tdnn   = (const float*)d_in[4];
    const float* b_tdnn   = (const float*)d_in[5];
    const float* bn_gamma = (const float*)d_in[6];
    const float* bn_beta  = (const float*)d_in[7];
    const float* w_conv   = (const float*)d_in[8];
    const float* b_conv   = (const float*)d_in[9];
    float* out = (float*)d_out;

    char* p = (char*)d_ws;
    auto alloc = [&](size_t bytes) {
        char* r = p;
        p += (bytes + 255) & ~(size_t)255;
        return r;
    };
    float* gmean          = (float*)alloc((size_t)BB * CC * 4);
    float* gstd           = (float*)alloc((size_t)BB * CC * 4);
    float* bias_val       = (float*)alloc((size_t)BB * CC * 4);
    float* bias_tdnn      = (float*)alloc((size_t)BB * AA * 4);
    unsigned short* wv_bf = (unsigned short*)alloc((size_t)CC * CC * 2);
    unsigned short* wt_bf = (unsigned short*)alloc((size_t)AA * CC * 2);
    unsigned short* wc_bf = (unsigned short*)alloc((size_t)CC * AA * 2);
    unsigned short* xT    = (unsigned short*)alloc((size_t)BB * LL * CC * 2);
    float4* partial       = (float4*)alloc((size_t)BB * NBLK * CC * 16);

    k_stats    <<<dim3(CC, BB),                256, 0, stream>>>(x, lengths, gmean, gstd);
    k_transpose<<<dim3(LL / 256, CC / 16, BB), 256, 0, stream>>>(x, xT);
    k_wconv    <<<dim3((CC * CC) / 256),       256, 0, stream>>>(w_val, w_tdnn, w_conv,
                                                                 wv_bf, wt_bf, wc_bf);
    k_bias     <<<dim3(BB),                    256, 0, stream>>>(gmean, gstd, w_val, b_val,
                                                                 w_tdnn, b_tdnn,
                                                                 bias_val, bias_tdnn);
    k_gemm     <<<dim3(NBLK, BB),              256, 0, stream>>>(xT, wv_bf, wt_bf, wc_bf,
                                                                 bias_val, bias_tdnn,
                                                                 bn_gamma, bn_beta, b_conv,
                                                                 lengths, partial);
    k_combine  <<<dim3(BB),                    256, 0, stream>>>(partial, out);
}
